// Gj_74921409511532
// MI455X (gfx1250) — matrix operand order repaired and checked by reading, not compiled, not run on hardware
//
#include <hip/hip_runtime.h>
#include <hip/hip_bf16.h>

typedef __attribute__((ext_vector_type(16))) _Float16 v16h;
typedef __attribute__((ext_vector_type(8)))  _Float16 v8h;
typedef __attribute__((ext_vector_type(4)))  _Float16 v4h;
typedef __attribute__((ext_vector_type(8)))  float    v8f;

#define O_DIM   256
#define NMAX_D  128
#define SPEC    4
#define ROWS_PB 32

// ---------------------------------------------------------------------------
// Kernel 1: zero species counters; transpose Ws [S,O,NMAX] f32 -> Wt [S,NMAX,O] f16
// so the WMMA B-operand (fixed N per lane, contiguous K) is a contiguous load.
// ---------------------------------------------------------------------------
__global__ void prep_kernel(const float* __restrict__ Ws,
                            _Float16* __restrict__ Wt,
                            int* __restrict__ cnt, int total) {
    int i = blockIdx.x * blockDim.x + threadIdx.x;
    if (i < SPEC) cnt[i] = 0;
    if (i < total) {
        int s = i / (O_DIM * NMAX_D);
        int r = i - s * (O_DIM * NMAX_D);
        int k = r / NMAX_D;          // row of W  (O dim)
        int n = r - k * NMAX_D;      // col of W  (NMAX dim)
        Wt[((size_t)s * NMAX_D + n) * O_DIM + k] = (_Float16)Ws[i];
    }
}

// ---------------------------------------------------------------------------
// Kernel 2: bucket atom indices by species (compaction -> 1x compute GEMMs).
// Bucket order is nondeterministic but per-atom output values are not.
// ---------------------------------------------------------------------------
__global__ void bucket_kernel(const int* __restrict__ species,
                              int* __restrict__ cnt,
                              int* __restrict__ buckets, int na) {
    int n = blockIdx.x * blockDim.x + threadIdx.x;
    if (n < na) {
        int s   = species[n];
        int pos = atomicAdd(&cnt[s], 1);
        buckets[s * na + pos] = n;
    }
}

// ---------------------------------------------------------------------------
// Kernel 3: per-species dense GEMM over compacted rows.
// Block: 256 threads (8 waves, wave32). Tile: 32 atoms x 128 outputs.
// Wave (wid) owns rows [wm, wm+16) x cols [wn, wn+32) -> 2 f32 16x16 accumulators,
// K loop of 8 steps of v_wmma_f32_16x16x32_f16.
// ---------------------------------------------------------------------------
__launch_bounds__(256, 2)
__global__ void gemm_kernel(const float* __restrict__ rho,
                            const float* __restrict__ bs,
                            const int* __restrict__ cnt,
                            const int* __restrict__ buckets,
                            const _Float16* __restrict__ Wt,
                            float* __restrict__ out, int na) {
    // +8 halves row pad: row stride 528B -> 16B-aligned, bank-conflict-free b128 reads
    __shared__ __align__(16) _Float16 As[ROWS_PB][O_DIM + 8];
    __shared__ int atomIdx[ROWS_PB];

    const int s     = blockIdx.y;
    const int m0    = blockIdx.x * ROWS_PB;
    const int count = cnt[s];
    if (m0 >= count) return;                         // uniform early-exit (over-launched grid)
    const int nvalid = min(ROWS_PB, count - m0);

    const int tid = threadIdx.x;

    // ---- stage A tile: gather rho rows via bucket indices, f32 -> f16 into LDS ----
    {
        const int row   = tid >> 3;                  // 0..31  (8 threads per row)
        const int cbase = (tid & 7) * 32;            // 32 floats per thread
        int atom = (row < nvalid) ? buckets[s * na + m0 + row] : -1;
        if ((tid & 7) == 0) atomIdx[row] = atom;
        if (atom >= 0) {
            const float4* src = (const float4*)(rho + (size_t)atom * O_DIM + cbase);
            #pragma unroll
            for (int j = 0; j < 8; ++j) {
                float4 v = src[j];
                v4h h; h[0] = (_Float16)v.x; h[1] = (_Float16)v.y;
                       h[2] = (_Float16)v.z; h[3] = (_Float16)v.w;
                *(v4h*)&As[row][cbase + j * 4] = h;
            }
        } else {
            v4h z = (v4h)(_Float16)0.0f;
            #pragma unroll
            for (int j = 0; j < 8; ++j) *(v4h*)&As[row][cbase + j * 4] = z;
        }
    }
    __syncthreads();

    // ---- per-wave WMMA compute ----
    const int wid  = tid >> 5;           // 0..7
    const int lane = tid & 31;
    const int half = lane >> 4;          // 0 | 1
    const int l16  = lane & 15;
    const int wm   = (wid >> 2) * 16;    // 0 | 16
    const int wn   = (wid & 3) * 32;     // 0,32,64,96

    // C init = bias broadcast (N depends only on lane -> all 8 rows share value)
    v8f acc0, acc1;
    {
        float b0 = bs[s * NMAX_D + wn + l16];
        float b1 = bs[s * NMAX_D + wn + 16 + l16];
        #pragma unroll
        for (int r = 0; r < 8; ++r) { acc0[r] = b0; acc1[r] = b1; }
    }

    const _Float16* wt_s = Wt + (size_t)s * NMAX_D * O_DIM;
    const int arow = wm + l16;

    #pragma unroll
    for (int k0 = 0; k0 < O_DIM; k0 += 32) {
        // A operand: contiguous K layout matching B. half0 lanes: K{k0..k0+15},
        // half1 lanes: K{k0+16..k0+31}. Two aligned ds_load_b128 each.
        const int kb = k0 + half * 16;
        v8h alo = *(const v8h*)&As[arow][kb];
        v8h ahi = *(const v8h*)&As[arow][kb + 8];
        v16h a;
        #pragma unroll
        for (int j = 0; j < 8; ++j) { a[j] = alo[j]; a[j + 8] = ahi[j]; }

        // B operand: lane<16 -> col=l16, K k0..k0+15 ; lane>=16 -> K k0+16..k0+31
        const int kb2 = k0 + half * 16;
        v16h b0m = *(const v16h*)(wt_s + (size_t)(wn +      l16) * O_DIM + kb2);
        v16h b1m = *(const v16h*)(wt_s + (size_t)(wn + 16 + l16) * O_DIM + kb2);

        acc0 = __builtin_amdgcn_wmma_f32_16x16x32_f16(false, a, false, b0m,
                                                      (short)0, acc0, false, false);
        acc1 = __builtin_amdgcn_wmma_f32_16x16x32_f16(false, a, false, b1m,
                                                      (short)0, acc1, false, false);
    }

    // ---- store: C layout = VGPR r -> M=r (lanes 0-15) / M=8+r (lanes 16-31) ----
    #pragma unroll
    for (int r = 0; r < 8; ++r) {
        int rt = wm + half * 8 + r;
        if (rt < nvalid) {
            float* dst = out + (size_t)atomIdx[rt] * NMAX_D + wn + l16;
            dst[0]  = acc0[r];
            dst[16] = acc1[r];
        }
    }
}

// ---------------------------------------------------------------------------
extern "C" void kernel_launch(void* const* d_in, const int* in_sizes, int n_in,
                              void* d_out, int out_size, void* d_ws, size_t ws_size,
                              hipStream_t stream) {
    const float* rho     = (const float*)d_in[0];   // [NA, 256] f32
    const float* Ws      = (const float*)d_in[1];   // [S, 256, 128] f32
    const float* bs      = (const float*)d_in[2];   // [S, 128] f32
    const int*   species = (const int*)d_in[3];     // [NA] i32
    float*       out     = (float*)d_out;           // [NA, 128] f32
    const int na = in_sizes[3];

    // scratch layout: [cnt: 4 ints][buckets: S*na ints][Wt: S*128*256 f16, 32B aligned]
    char* ws = (char*)d_ws;
    int* cnt     = (int*)ws;
    int* buckets = (int*)(ws + 16);
    size_t wt_off = (16 + (size_t)SPEC * na * sizeof(int) + 31) & ~(size_t)31;
    _Float16* Wt = (_Float16*)(ws + wt_off);

    const int totalW = SPEC * O_DIM * NMAX_D;
    prep_kernel<<<(totalW + 255) / 256, 256, 0, stream>>>(Ws, Wt, cnt, totalW);
    bucket_kernel<<<(na + 255) / 256, 256, 0, stream>>>(species, cnt, buckets, na);

    dim3 grid((na + ROWS_PB - 1) / ROWS_PB, SPEC);
    gemm_kernel<<<grid, 256, 0, stream>>>(rho, bs, cnt, buckets, Wt, out, na);
}
